// GPSLayer_41077067219433
// MI455X (gfx1250) — matrix operand order repaired and checked by reading, not compiled, not run on hardware
//
#include <hip/hip_runtime.h>
#include <hip/hip_bf16.h>
#include <math.h>

#define N_NODES 4096
#define DIM     256
#define EDIM    64
#define NE      262144
#define NH      8
#define HDIM    32

typedef __attribute__((ext_vector_type(16))) __bf16 v16bf;
typedef __attribute__((ext_vector_type(8)))  float  v8f;

union Frag16 {
  v16bf v;
  __bf16 e[16];
  unsigned w[8];
};

__device__ __forceinline__ v8f vzero8() {
  v8f z;
#pragma unroll
  for (int i = 0; i < 8; i++) z[i] = 0.0f;
  return z;
}

__device__ __forceinline__ v8f wmma_bf16(const Frag16& a, const Frag16& b, v8f c) {
  return __builtin_amdgcn_wmma_f32_16x16x32_bf16(false, a.v, false, b.v,
                                                 (short)0, c, false, false);
}

// pack two floats into one dword of bf16x2 (lowers to v_cvt_pk_bf16_f32)
__device__ __forceinline__ unsigned pk_bf16(float a, float b) {
  union { __bf16 h[2]; unsigned u; } t;
  t.h[0] = (__bf16)a;
  t.h[1] = (__bf16)b;
  return t.u;
}

struct uint2_ { unsigned x, y; };

// cvt float4 -> 2 dwords of packed bf16
__device__ __forceinline__ uint2_ pk4_bf16(float4 f) {
  uint2_ r;
  r.x = pk_bf16(f.x, f.y);
  r.y = pk_bf16(f.z, f.w);
  return r;
}

// ---------------------------------------------------------------------------
// GINEConv edge stage: msg = relu(x[src] + edge_attr @ Wl + bl); agg[dst] += msg
// Block = 256 threads (8 waves), 128 edges per block. Wl staged once in LDS.
// ---------------------------------------------------------------------------
__global__ __launch_bounds__(256) void edge_gine_kernel(
    const float* __restrict__ x, const long long* __restrict__ ei,
    const float* __restrict__ ea, const float* __restrict__ Wl,
    const float* __restrict__ bl, float* __restrict__ agg) {
  __shared__ __bf16 Ws[DIM][EDIM];   // Ws[n][k] = Wl[k][n], 32 KB
  __shared__ __bf16 Es[128][EDIM];   // edge_attr tile, 16 KB

  const int tid = threadIdx.x;
  const int wv = tid >> 5, lane = tid & 31;
  const int lh = lane >> 4, l16 = lane & 15;
  const long e0 = (long)blockIdx.x * 128;

  // stage Wl (b128 loads coalesced over n, transposed scalar LDS stores)
  for (int i = tid; i < DIM * EDIM / 4; i += 256) {
    int k = i >> 6, n4 = (i & 63) * 4;
    float4 f = *(const float4*)&Wl[k * DIM + n4];
    Ws[n4 + 0][k] = (__bf16)f.x;
    Ws[n4 + 1][k] = (__bf16)f.y;
    Ws[n4 + 2][k] = (__bf16)f.z;
    Ws[n4 + 3][k] = (__bf16)f.w;
  }
  // stage edge_attr tile (b128 loads, packed cvt, b64 LDS stores)
  for (int i = tid; i < 128 * EDIM / 4; i += 256) {
    int r = i >> 4, c4 = (i & 15) * 4;
    float4 f = *(const float4*)&ea[(e0 + r) * EDIM + c4];
    *(uint2_*)&Es[r][c4] = pk4_bf16(f);
  }
  __syncthreads();

  // A fragments for this wave's 16 edges, K=64 -> two k-steps
  Frag16 af[2];
  {
    const __bf16* rp = &Es[wv * 16 + l16][0];
#pragma unroll
    for (int ks = 0; ks < 2; ks++)
#pragma unroll
      for (int p = 0; p < 8; p++) {
        int kk = ks * 32 + 16 * lh + 2 * p;
        af[ks].w[p] = *(const unsigned*)(rp + kk);
      }
  }

  long long srcs[8], dsts[8];
#pragma unroll
  for (int v = 0; v < 8; v++) {
    long e = e0 + wv * 16 + v + 8 * lh;
    srcs[v] = ei[e];
    dsts[v] = ei[(long)NE + e];
  }

#pragma unroll 1
  for (int nt = 0; nt < 16; nt++) {
    v8f acc = vzero8();
    const __bf16* rp = &Ws[nt * 16 + l16][0];
#pragma unroll
    for (int ks = 0; ks < 2; ks++) {
      Frag16 bf;
#pragma unroll
      for (int p = 0; p < 8; p++)
        bf.w[p] = *(const unsigned*)(rp + ks * 32 + 16 * lh + 2 * p);
      acc = wmma_bf16(af[ks], bf, acc);
    }
    const int c = nt * 16 + l16;
    const float blc = bl[c];
    // issue all gathers first so they overlap, then the atomics
    float xs[8];
#pragma unroll
    for (int v = 0; v < 8; v++) xs[v] = x[srcs[v] * DIM + c];
#pragma unroll
    for (int v = 0; v < 8; v++) {
      float m = fmaxf(acc[v] + blc + xs[v], 0.0f);
      unsafeAtomicAdd(agg + dsts[v] * DIM + c, m);   // global_atomic_add_f32
    }
  }
}

// ---------------------------------------------------------------------------
// h = (1+eps)*x + agg
// ---------------------------------------------------------------------------
__global__ void gine_combine(const float* __restrict__ x, const float* __restrict__ agg,
                             const float* __restrict__ eps, float* __restrict__ h, int n) {
  const float c = 1.0f + eps[0];
  for (int i = blockIdx.x * blockDim.x + threadIdx.x; i < n; i += gridDim.x * blockDim.x)
    h[i] = c * x[i] + agg[i];
}

// ---------------------------------------------------------------------------
// Generic bf16-WMMA GEMM: OUT[M,N] = act(A[M,K] @ B + bias[N])
// BT=false: B is [K,N] row-major.  BT=true: B is [N,K] row-major (x @ W.T).
// EPI: 0=none, 1=ELU, 2=exact GELU.
// Block 128 threads (4 waves), tile 64x64, K-step 32.
// ---------------------------------------------------------------------------
template <int EPI, bool BT>
__global__ __launch_bounds__(128) void gemm_wmma(
    const float* __restrict__ A, const float* __restrict__ B,
    const float* __restrict__ bias, float* __restrict__ OUT,
    int M, int N, int K) {
  __shared__ __bf16 As[64][32];
  __shared__ __bf16 Bs[64][32];   // Bs[n][k]

  const int tid = threadIdx.x;
  const int wv = tid >> 5, lane = tid & 31;
  const int lh = lane >> 4, l16 = lane & 15;
  const int wm = (wv >> 1) * 32, wn = (wv & 1) * 32;
  const int bm = blockIdx.y * 64, bn = blockIdx.x * 64;

  v8f acc[2][2];
#pragma unroll
  for (int mt = 0; mt < 2; mt++)
#pragma unroll
    for (int nt = 0; nt < 2; nt++) acc[mt][nt] = vzero8();

  for (int k0 = 0; k0 < K; k0 += 32) {
    // A tile: 64x32 f32 = 512 float4, b128 loads + packed cvt + b64 LDS stores
    for (int i = tid; i < 512; i += 128) {
      int r = i >> 3, c4 = (i & 7) * 4;
      float4 f = *(const float4*)&A[(long)(bm + r) * K + (k0 + c4)];
      *(uint2_*)&As[r][c4] = pk4_bf16(f);
    }
    if (k0 + 32 < K)   // prefetch next A tile (global_prefetch_b8)
      __builtin_prefetch(&A[(long)(bm + (tid >> 3)) * K + (k0 + 32)], 0, 1);
    if (BT) {
      for (int i = tid; i < 512; i += 128) {
        int n = i >> 3, k4 = (i & 7) * 4;
        float4 f = *(const float4*)&B[(long)(bn + n) * K + (k0 + k4)];
        *(uint2_*)&Bs[n][k4] = pk4_bf16(f);
      }
    } else {
      for (int i = tid; i < 512; i += 128) {
        int k = i >> 4, n4 = (i & 15) * 4;
        float4 f = *(const float4*)&B[(long)(k0 + k) * N + (bn + n4)];
        Bs[n4 + 0][k] = (__bf16)f.x;
        Bs[n4 + 1][k] = (__bf16)f.y;
        Bs[n4 + 2][k] = (__bf16)f.z;
        Bs[n4 + 3][k] = (__bf16)f.w;
      }
    }
    __syncthreads();

    Frag16 af[2], bf[2];
#pragma unroll
    for (int mt = 0; mt < 2; mt++) {
      const __bf16* rp = &As[wm + mt * 16 + l16][0];
#pragma unroll
      for (int p = 0; p < 8; p++) {
        int kk = 16 * lh + 2 * p;
        af[mt].w[p] = *(const unsigned*)(rp + kk);
      }
    }
#pragma unroll
    for (int nt = 0; nt < 2; nt++) {
      const __bf16* rp = &Bs[wn + nt * 16 + l16][0];
#pragma unroll
      for (int p = 0; p < 8; p++)
        bf[nt].w[p] = *(const unsigned*)(rp + 16 * lh + 2 * p);
    }
#pragma unroll
    for (int mt = 0; mt < 2; mt++)
#pragma unroll
      for (int nt = 0; nt < 2; nt++)
        acc[mt][nt] = wmma_bf16(af[mt], bf[nt], acc[mt][nt]);
    __syncthreads();
  }

#pragma unroll
  for (int mt = 0; mt < 2; mt++)
#pragma unroll
    for (int nt = 0; nt < 2; nt++) {
      const int c = bn + wn + nt * 16 + l16;
      const float bs = bias[c];
#pragma unroll
      for (int v = 0; v < 8; v++) {
        const int r = bm + wm + mt * 16 + v + 8 * lh;
        float xv = acc[mt][nt][v] + bs;
        if (EPI == 1) xv = xv > 0.0f ? xv : (expf(xv) - 1.0f);
        else if (EPI == 2) xv = 0.5f * xv * (1.0f + erff(xv * 0.70710678118654752f));
        OUT[(long)r * N + c] = xv;
      }
    }
}

// ---------------------------------------------------------------------------
// Flash attention: block = 16 queries, 8 waves = 8 heads. HD=32. bf16 WMMA,
// fp32 accumulators, online softmax.
// ---------------------------------------------------------------------------
__global__ __launch_bounds__(256) void attn_kernel(
    const float* __restrict__ qkv, float* __restrict__ o) {
  __shared__ __bf16 Kl[NH][32][32];   // [head][key][hd]   16 KB
  __shared__ __bf16 Vl[NH][32][32];   // [head][hd][key]   16 KB
  __shared__ __bf16 Pl[NH][16][32];   // [head][row][key]   8 KB

  const int tid = threadIdx.x;
  const int h = tid >> 5, lane = tid & 31;
  const int lh = lane >> 4, l16 = lane & 15;
  const int q0 = blockIdx.x * 16;
  const float scale = 0.17677669529663687f;   // 1/sqrt(32)

  Frag16 qf;
  {
    const float* qrow = qkv + (long)(q0 + l16) * (3 * DIM) + h * HDIM;
#pragma unroll
    for (int j = 0; j < 16; j++) {
      int kk = 16 * lh + j;
      qf.e[j] = (__bf16)qrow[kk];
    }
  }

  float m[8], l[8];
  v8f accO[2];
#pragma unroll
  for (int v = 0; v < 8; v++) { m[v] = -1e30f; l[v] = 0.0f; }
  accO[0] = vzero8();
  accO[1] = vzero8();

  for (int kc = 0; kc < N_NODES / 32; kc++) {
    const int kb = kc * 32;
    __syncthreads();
    // stage K (b128 + packed cvt + b64 store) and V (b128 + transposed scalar)
    for (int i = lane; i < 256; i += 32) {
      int key = i >> 3, d4 = (i & 7) * 4;
      const float* kr = qkv + (long)(kb + key) * (3 * DIM) + h * HDIM;
      float4 fk = *(const float4*)&kr[DIM + d4];
      *(uint2_*)&Kl[h][key][d4] = pk4_bf16(fk);
      float4 fv = *(const float4*)&kr[2 * DIM + d4];
      Vl[h][d4 + 0][key] = (__bf16)fv.x;
      Vl[h][d4 + 1][key] = (__bf16)fv.y;
      Vl[h][d4 + 2][key] = (__bf16)fv.z;
      Vl[h][d4 + 3][key] = (__bf16)fv.w;
    }
    __syncthreads();

    v8f S[2];
#pragma unroll
    for (int t = 0; t < 2; t++) {
      Frag16 bfK;
      const __bf16* rp = &Kl[h][t * 16 + l16][0];
#pragma unroll
      for (int p = 0; p < 8; p++)
        bfK.w[p] = *(const unsigned*)(rp + 16 * lh + 2 * p);
      S[t] = wmma_bf16(qf, bfK, vzero8());
    }

#pragma unroll
    for (int v = 0; v < 8; v++) {
      S[0][v] *= scale;
      S[1][v] *= scale;
      float rm = fmaxf(S[0][v], S[1][v]);
#pragma unroll
      for (int off = 8; off > 0; off >>= 1) rm = fmaxf(rm, __shfl_xor(rm, off, 32));
      const float mn = fmaxf(m[v], rm);
      const float alpha = expf(m[v] - mn);
      m[v] = mn;
      const float p0 = expf(S[0][v] - mn);
      const float p1 = expf(S[1][v] - mn);
      float rs = p0 + p1;
#pragma unroll
      for (int off = 8; off > 0; off >>= 1) rs += __shfl_xor(rs, off, 32);
      l[v] = l[v] * alpha + rs;
      Pl[h][v + 8 * lh][l16]      = (__bf16)p0;
      Pl[h][v + 8 * lh][16 + l16] = (__bf16)p1;
      accO[0][v] *= alpha;
      accO[1][v] *= alpha;
    }
    __syncthreads();

    Frag16 pf;
    {
      const __bf16* rp = &Pl[h][l16][0];
#pragma unroll
      for (int p = 0; p < 8; p++) {
        int kk = 16 * lh + 2 * p;
        pf.w[p] = *(const unsigned*)(rp + kk);
      }
    }
#pragma unroll
    for (int nt = 0; nt < 2; nt++) {
      Frag16 vf;
      const __bf16* rp = &Vl[h][nt * 16 + l16][0];
#pragma unroll
      for (int p = 0; p < 8; p++)
        vf.w[p] = *(const unsigned*)(rp + 16 * lh + 2 * p);
      accO[nt] = wmma_bf16(pf, vf, accO[nt]);
    }
  }

#pragma unroll
  for (int nt = 0; nt < 2; nt++)
#pragma unroll
    for (int v = 0; v < 8; v++) {
      const int node = q0 + v + 8 * lh;
      o[(long)node * DIM + h * HDIM + nt * 16 + l16] = accO[nt][v] / l[v];
    }
}

// ---------------------------------------------------------------------------
// out = res + [elu](LayerNorm(in) * g + b), one row per block, D=256
// ---------------------------------------------------------------------------
template <bool ELU_OUT>
__global__ __launch_bounds__(256) void ln_residual(
    const float* __restrict__ in, const float* __restrict__ res,
    const float* __restrict__ g, const float* __restrict__ b,
    float* __restrict__ out) {
  const int row = blockIdx.x, t = threadIdx.x;
  const int wv = t >> 5, lane = t & 31;
  __shared__ float red[8];
  __shared__ float bmean, brstd;

  const float v = in[(long)row * DIM + t];
  float s = v;
#pragma unroll
  for (int off = 16; off > 0; off >>= 1) s += __shfl_xor(s, off, 32);
  if (lane == 0) red[wv] = s;
  __syncthreads();
  if (t == 0) {
    float tot = 0.0f;
    for (int i = 0; i < 8; i++) tot += red[i];
    bmean = tot / DIM;
  }
  __syncthreads();
  const float d = v - bmean;
  float s2 = d * d;
#pragma unroll
  for (int off = 16; off > 0; off >>= 1) s2 += __shfl_xor(s2, off, 32);
  if (lane == 0) red[wv] = s2;
  __syncthreads();
  if (t == 0) {
    float tot = 0.0f;
    for (int i = 0; i < 8; i++) tot += red[i];
    brstd = rsqrtf(tot / DIM + 1e-5f);
  }
  __syncthreads();
  float y = d * brstd * g[t] + b[t];
  if (ELU_OUT) y = y > 0.0f ? y : (expf(y) - 1.0f);
  out[(long)row * DIM + t] = res[(long)row * DIM + t] + y;
}

// ---------------------------------------------------------------------------
extern "C" void kernel_launch(void* const* d_in, const int* in_sizes, int n_in,
                              void* d_out, int out_size, void* d_ws, size_t ws_size,
                              hipStream_t stream) {
  (void)in_sizes; (void)n_in; (void)out_size; (void)ws_size;
  const float*     x     = (const float*)d_in[0];
  const long long* ei    = (const long long*)d_in[1];
  const float*     ea    = (const float*)d_in[2];
  const float*     eps   = (const float*)d_in[3];
  const float*     Wl    = (const float*)d_in[4];
  const float*     bl    = (const float*)d_in[5];
  const float*     W1    = (const float*)d_in[6];
  const float*     b1    = (const float*)d_in[7];
  const float*     W2    = (const float*)d_in[8];
  const float*     b2    = (const float*)d_in[9];
  const float*     lng   = (const float*)d_in[10];
  const float*     lnb   = (const float*)d_in[11];
  const float*     in_w  = (const float*)d_in[12];
  const float*     in_b  = (const float*)d_in[13];
  const float*     out_w = (const float*)d_in[14];
  const float*     out_b = (const float*)d_in[15];
  const float*     lag   = (const float*)d_in[16];
  const float*     lab   = (const float*)d_in[17];
  const float*     Wf1   = (const float*)d_in[18];
  const float*     bf1   = (const float*)d_in[19];
  const float*     Wf2   = (const float*)d_in[20];
  const float*     bf2   = (const float*)d_in[21];
  const float*     lfg   = (const float*)d_in[22];
  const float*     lfb   = (const float*)d_in[23];

  float* ws = (float*)d_ws;
  const size_t ND = (size_t)N_NODES * DIM;
  float* agg = ws;             // ND
  float* h   = ws + 1 * ND;    // ND
  float* t1  = ws + 2 * ND;    // ND
  float* h2  = ws + 3 * ND;    // ND
  float* x1  = ws + 4 * ND;    // ND
  float* qkv = ws + 5 * ND;    // 3*ND
  float* oat = ws + 8 * ND;    // ND
  float* opj = ws + 9 * ND;    // ND
  float* x2  = ws + 10 * ND;   // ND
  float* f1  = ws + 11 * ND;   // 2*ND
  float* f2  = ws + 13 * ND;   // ND
  float* out = (float*)d_out;

  // --- GINEConv ---
  hipMemsetAsync(agg, 0, ND * sizeof(float), stream);
  edge_gine_kernel<<<NE / 128, 256, 0, stream>>>(x, ei, ea, Wl, bl, agg);
  gine_combine<<<512, 256, 0, stream>>>(x, agg, eps, h, (int)ND);
  gemm_wmma<1, false><<<dim3(DIM / 64, N_NODES / 64), 128, 0, stream>>>(h, W1, b1, t1, N_NODES, DIM, DIM);
  gemm_wmma<0, false><<<dim3(DIM / 64, N_NODES / 64), 128, 0, stream>>>(t1, W2, b2, h2, N_NODES, DIM, DIM);
  ln_residual<true><<<N_NODES, 256, 0, stream>>>(h2, x, lng, lnb, x1);
  // --- MHA ---
  gemm_wmma<0, true><<<dim3(3 * DIM / 64, N_NODES / 64), 128, 0, stream>>>(x1, in_w, in_b, qkv, N_NODES, 3 * DIM, DIM);
  attn_kernel<<<N_NODES / 16, 256, 0, stream>>>(qkv, oat);
  gemm_wmma<0, true><<<dim3(DIM / 64, N_NODES / 64), 128, 0, stream>>>(oat, out_w, out_b, opj, N_NODES, DIM, DIM);
  ln_residual<false><<<N_NODES, 256, 0, stream>>>(opj, x1, lag, lab, x2);
  // --- FFN ---
  gemm_wmma<2, false><<<dim3(2 * DIM / 64, N_NODES / 64), 128, 0, stream>>>(x2, Wf1, bf1, f1, N_NODES, 2 * DIM, DIM);
  gemm_wmma<0, false><<<dim3(DIM / 64, N_NODES / 64), 128, 0, stream>>>(f1, Wf2, bf2, f2, N_NODES, DIM, 2 * DIM);
  ln_residual<false><<<N_NODES, 256, 0, stream>>>(f2, x2, lfg, lfb, out);
}
